// SS2D_44822278701564
// MI455X (gfx1250) — matrix operand order repaired and checked by reading, not compiled, not run on hardware
//
#include <hip/hip_runtime.h>

// ---------------- problem constants (from reference setup_inputs) ------------
#define B_   8
#define H_   48
#define W_   48
#define DM   96      // model dim
#define DIN  192     // inner dim
#define NS   16      // state dim N
#define RR   6       // dt rank
#define KK   4       // scan directions
#define L_   (H_ * W_)          // 2304
#define M1   (B_ * L_)          // 18432 rows for all GEMMs
#define CPAD 48                 // R + 2N = 38 padded to 48 (3 N-tiles of 16)

typedef __bf16 bf16;
typedef __attribute__((ext_vector_type(16))) __bf16 v16bf;
typedef __attribute__((ext_vector_type(8)))  __bf16 v8bf;
typedef __attribute__((ext_vector_type(8)))  float  v8f;

// ---------------------------------------------------------------------------
// bf16 WMMA GEMM: C(MxN,f32) = A(MxK,bf16 row-major) * B(KxN, pre-packed).
// One wave32 computes a 16 x (16*NT) strip: one A fragment feeds NT
// v_wmma_f32_16x16x32_bf16 issues per 32-wide K-step.
// B is pre-packed in fragment order [ntile][kstep][lane][16 halves], so each
// WMMA's B operand is a single contiguous 32-byte load.
// Fragment layouts per CDNA5 ISA:
//   A 16x32: lanes 0-15 row M=lane, K {0..7}U{16..23}; lanes 16-31 {8..15}U{24..31}
//   B 32x16: lanes 0-15 col N=lane, K {0..7}U{16..23}; lanes 16-31 K {8..15}U{24..31}
//   C 16x16 f32: acc[r] -> row (r + 8*hi), col = lane&15
// ---------------------------------------------------------------------------
template <int NT>
__global__ void gemm_bf16_wmma(const bf16* __restrict__ A,
                               const bf16* __restrict__ Bpk,
                               float* __restrict__ C,
                               int M, int N, int K) {
  int gtid = blockIdx.x * blockDim.x + threadIdx.x;
  int wave = gtid >> 5;
  int lane = gtid & 31;
  int ngroups = (N >> 4) / NT;
  int mtiles  = M >> 4;
  if (wave >= mtiles * ngroups) return;     // whole-wave uniform exit
  int mt  = wave / ngroups;
  int ng  = wave % ngroups;
  int r16 = lane & 15;
  int hi  = lane >> 4;
  int row = (mt << 4) + r16;
  int ksteps = K >> 5;

  v8f acc[NT] = {};
  for (int ks = 0; ks < ksteps; ++ks) {
    int k0 = ks << 5;
    const bf16* ap = A + (size_t)row * K + k0 + hi * 8;
    v8bf alo = *(const v8bf*)ap;            // 16B load
    v8bf ahi = *(const v8bf*)(ap + 16);     // 16B load
    v16bf a = __builtin_shufflevector(alo, ahi,
              0, 1, 2, 3, 4, 5, 6, 7, 8, 9, 10, 11, 12, 13, 14, 15);
#pragma unroll
    for (int t = 0; t < NT; ++t) {
      int nt = ng * NT + t;
      const v16bf b =
          *(const v16bf*)(Bpk + (((size_t)nt * ksteps + ks) * 32 + lane) * 16);
      acc[t] = __builtin_amdgcn_wmma_f32_16x16x32_bf16(
          /*neg_a=*/false, a, /*neg_b=*/false, b,
          /*c_mod=*/(short)0, acc[t], /*reuse_a=*/false, /*reuse_b=*/false);
    }
  }
#pragma unroll
  for (int t = 0; t < NT; ++t) {
    int col = (ng * NT + t) * 16 + r16;
    float* cp = C + (size_t)((mt << 4) + hi * 8) * N + col;
#pragma unroll
    for (int r = 0; r < 8; ++r) cp[(size_t)r * N] = acc[t][r];
  }
}

// ------------------------- small prep kernels ------------------------------
__global__ void k_f32_to_bf16(const float* __restrict__ in, bf16* __restrict__ out, int n) {
  int i = blockIdx.x * blockDim.x + threadIdx.x;
  if (i < n) out[i] = (bf16)in[i];
}

// Pack an (N x K) f32 row-major weight (rows = output channels) into WMMA
// B-fragment order: dst[((nt*ksteps + ks)*32 + lane)*16 + e] =
//   src[n*K + kk], n = nt*16 + (lane&15), kk = ks*32 + (lane>>4)*8 + (e&7) + (e>>3)*16.
// Rows n >= Nvalid are zero-padded (used for the 38->48 x_proj pad).
__global__ void k_pack_bfrag(const float* __restrict__ src, bf16* __restrict__ dst,
                             int N, int K, int Nvalid) {
  int i = blockIdx.x * blockDim.x + threadIdx.x;
  int ksteps = K >> 5;
  int total = (N >> 4) * ksteps * 512;
  if (i >= total) return;
  int e    = i & 15;
  int lane = (i >> 4) & 31;
  int ks   = (i >> 9) % ksteps;
  int nt   = (i >> 9) / ksteps;
  int n  = (nt << 4) + (lane & 15);
  int kk = (ks << 5) + (lane >> 4) * 8 + (e & 7) + (e >> 3) * 16;
  float v = (n < Nvalid) ? src[(size_t)n * K + kk] : 0.f;
  dst[i] = (bf16)v;
}

// ---------------- depthwise 3x3 conv (SAME) + SiLU -------------------------
// reads xc = xz[..., 0:192]; writes channel-major f32 (scan u) + spatial bf16
__global__ void k_conv_silu(const float* __restrict__ xz, const float* __restrict__ cw,
                            const float* __restrict__ cb,
                            float* __restrict__ xconv, bf16* __restrict__ xsp) {
  int i = blockIdx.x * blockDim.x + threadIdx.x;
  if (i >= B_ * L_ * DIN) return;
  int d = i % DIN;
  int l = (i / DIN) % L_;
  int b = i / (DIN * L_);
  int h = l / W_, w = l % W_;
  float acc = cb[d];
#pragma unroll
  for (int kh = 0; kh < 3; ++kh) {
    int hh = h + kh - 1;
    if (hh < 0 || hh >= H_) continue;
#pragma unroll
    for (int kw = 0; kw < 3; ++kw) {
      int ww = w + kw - 1;
      if (ww < 0 || ww >= W_) continue;
      acc += cw[d * 9 + kh * 3 + kw] * xz[((size_t)(b * L_ + hh * W_ + ww)) * 384 + d];
    }
  }
  float s = acc / (1.f + __expf(-acc));                 // SiLU
  xconv[((size_t)b * DIN + d) * L_ + l] = s;            // channel-major (scan)
  xsp[((size_t)(b * L_) + l) * DIN + d] = (bf16)s;      // spatial-major (GEMM A)
}

// ---------------------- selective scan (serial core) -----------------------
// One block per (b,k), 192 threads = one d-channel each (6 wave32s).
// Per step: stage the 48 projection values (dt[6] | B[16] | C[16] | pad) into
// LDS once, broadcast to all 192 threads; fp32 recurrence with N=16 state in
// registers.  ys stored [b*L+s][k][d] so the merge kernel reads contiguously.
__global__ void k_scan(const float* __restrict__ xconv, const float* __restrict__ xdbl,
                       const float* __restrict__ dtw, const float* __restrict__ dtb,
                       const float* __restrict__ A_logs, const float* __restrict__ Ds,
                       float* __restrict__ ys) {
  int b = blockIdx.x >> 2;
  int k = blockIdx.x & 3;
  int d = threadIdx.x;                      // 0..191
  __shared__ __align__(16) float sd[CPAD];

  int kd = k * DIN + d;
  float A[NS];
#pragma unroll
  for (int nn = 0; nn < NS; ++nn) A[nn] = -__expf(A_logs[(size_t)kd * NS + nn]);
  float wdt[RR];
#pragma unroll
  for (int r = 0; r < RR; ++r) wdt[r] = dtw[(size_t)kd * RR + r];
  float bias = dtb[kd];
  float Dv   = Ds[kd];
  float h[NS];
#pragma unroll
  for (int nn = 0; nn < NS; ++nn) h[nn] = 0.f;

  const float* xc = xconv + ((size_t)b * DIN + d) * L_;
  const float* xd = xdbl + ((size_t)k * M1 + (size_t)b * L_) * CPAD;
  float* yo = ys + (size_t)b * L_ * KK * DIN + (size_t)k * DIN + d;

  for (int i = 0; i < L_; ++i) {
    // traversal: k&1 -> column-major transpose, k&2 -> reversed sequence
    int t = (k & 2) ? (L_ - 1 - i) : i;
    int s = (k & 1) ? ((t % H_) * W_ + (t / H_)) : t;
    __syncthreads();
    if (d < CPAD / 4)
      ((float4*)sd)[d] = ((const float4*)(xd + (size_t)s * CPAD))[d];
    __syncthreads();
    if (d == 0 && i + 1 < L_) {             // prefetch next step's row
      int t2 = (k & 2) ? (L_ - 2 - i) : (i + 1);
      int s2 = (k & 1) ? ((t2 % H_) * W_ + (t2 / H_)) : t2;
      __builtin_prefetch(xd + (size_t)s2 * CPAD, 0, 1);
    }
    float u  = xc[s];
    float dp = bias;
#pragma unroll
    for (int r = 0; r < RR; ++r) dp += wdt[r] * sd[r];
    float delta = (dp > 20.f) ? dp : log1pf(__expf(dp));  // softplus
    float du = delta * u;
    float y  = 0.f;
#pragma unroll
    for (int nn = 0; nn < NS; ++nn) {
      h[nn] = h[nn] * __expf(delta * A[nn]) + du * sd[RR + nn];   // h*dA + dB*u
      y += h[nn] * sd[RR + NS + nn];                               // C . h
    }
    yo[(size_t)s * KK * DIN] = y + Dv * u;   // + Ds skip, scattered to spatial s
  }
}

// ------------- CrossMerge + LayerNorm + SiLU gate -> bf16 ------------------
// One block per pixel (b,l), 192 threads over channels.
__global__ void k_merge_ln_gate(const float* __restrict__ ys, const float* __restrict__ xz,
                                const float* __restrict__ lnw, const float* __restrict__ lnb,
                                bf16* __restrict__ yg) {
  int m = blockIdx.x;          // b*L + l
  int d = threadIdx.x;         // 0..191
  const float* yr = ys + (size_t)m * KK * DIN;
  float v = yr[d] + yr[DIN + d] + yr[2 * DIN + d] + yr[3 * DIN + d];

  __shared__ float red[DIN];
  red[d] = v;
  __syncthreads();
#pragma unroll
  for (int off = 96; off >= 6; off >>= 1) {
    if (d < off) red[d] += red[d + off];
    __syncthreads();
  }
  if (d == 0) red[0] = (red[0] + red[1] + red[2] + red[3] + red[4] + red[5]) * (1.f / DIN);
  __syncthreads();
  float mu = red[0];
  __syncthreads();
  float c = v - mu;
  red[d] = c * c;
  __syncthreads();
#pragma unroll
  for (int off = 96; off >= 6; off >>= 1) {
    if (d < off) red[d] += red[d + off];
    __syncthreads();
  }
  if (d == 0) red[0] = (red[0] + red[1] + red[2] + red[3] + red[4] + red[5]) * (1.f / DIN);
  __syncthreads();
  float var = red[0];

  float yln = c * rsqrtf(var + 1e-5f) * lnw[d] + lnb[d];
  float z = xz[(size_t)m * 384 + DIN + d];          // gate branch
  float g = z / (1.f + __expf(-z));                 // SiLU
  yg[(size_t)m * DIN + d] = (bf16)(yln * g);
}

// ---------------------------------------------------------------------------
extern "C" void kernel_launch(void* const* d_in, const int* in_sizes, int n_in,
                              void* d_out, int out_size, void* d_ws, size_t ws_size,
                              hipStream_t stream) {
  (void)in_sizes; (void)n_in; (void)out_size; (void)ws_size;
  const float* x_in    = (const float*)d_in[0];   // (B,H,W,96)
  const float* inp_w   = (const float*)d_in[1];   // (384,96)   = (N,K)
  const float* conv_w  = (const float*)d_in[2];   // (192,1,3,3)
  const float* conv_b  = (const float*)d_in[3];   // (192,)
  const float* xproj_w = (const float*)d_in[4];   // (4,38,192) = 4 x (N=38,K=192)
  const float* dt_w    = (const float*)d_in[5];   // (4,192,6)
  const float* dt_b    = (const float*)d_in[6];   // (4,192)
  const float* A_logs  = (const float*)d_in[7];   // (768,16)
  const float* Ds      = (const float*)d_in[8];   // (768,)
  const float* ln_w    = (const float*)d_in[9];   // (192,)
  const float* ln_b    = (const float*)d_in[10];  // (192,)
  const float* out_w   = (const float*)d_in[11];  // (96,192)   = (N,K)
  float* out = (float*)d_out;                     // (B,H,W,96)

  // ------------------ workspace layout (256B aligned) ----------------------
  char* ws = (char*)d_ws;
  size_t off = 0;
  auto alloc = [&](size_t bytes) -> char* {
    char* p = ws + off;
    off += (bytes + 255) & ~(size_t)255;
    return p;
  };
  float* xz    = (float*)alloc((size_t)M1 * 384 * 4);          // in_proj out
  float* xconv = (float*)alloc((size_t)B_ * DIN * L_ * 4);     // conv f32 (chan-major)
  bf16*  xsp   = (bf16*) alloc((size_t)M1 * DIN * 2);          // conv bf16 (spatial)
  float* xdbl  = (float*)alloc((size_t)KK * M1 * CPAD * 4);    // dt|B|C projections
  float* ys    = (float*)alloc((size_t)M1 * KK * DIN * 4);     // scan outputs
  bf16*  yg    = (bf16*) alloc((size_t)M1 * DIN * 2);          // gated, pre out_proj
  bf16*  xbf   = (bf16*) alloc((size_t)M1 * DM * 2);           // x as bf16
  bf16*  w1pk  = (bf16*) alloc((size_t)384 * DM * 2);          // in_proj packed frags
  bf16*  wxpk  = (bf16*) alloc((size_t)KK * CPAD * DIN * 2);   // x_proj packed frags
  bf16*  wopk  = (bf16*) alloc((size_t)DM * DIN * 2);          // out_proj packed frags

  int n;
  // input conversion + weight fragment packing
  n = M1 * DM;
  k_f32_to_bf16<<<(n + 255) / 256, 256, 0, stream>>>(x_in, xbf, n);
  n = (384 / 16) * (DM / 32) * 512;
  k_pack_bfrag<<<(n + 255) / 256, 256, 0, stream>>>(inp_w, w1pk, 384, DM, 384);
  n = (DM / 16) * (DIN / 32) * 512;
  k_pack_bfrag<<<(n + 255) / 256, 256, 0, stream>>>(out_w, wopk, DM, DIN, DM);
  n = (CPAD / 16) * (DIN / 32) * 512;
  for (int k = 0; k < KK; ++k)
    k_pack_bfrag<<<(n + 255) / 256, 256, 0, stream>>>(
        xproj_w + (size_t)k * (RR + 2 * NS) * DIN,
        wxpk + (size_t)k * CPAD * DIN, CPAD, DIN, RR + 2 * NS);

  // 1) in_proj: xz(18432x384) = x(18432x96) * W^T(96x384)   [WMMA, NT=4]
  {
    int waves = (M1 / 16) * ((384 / 16) / 4);
    int thr = waves * 32;
    gemm_bf16_wmma<4><<<(thr + 255) / 256, 256, 0, stream>>>(xbf, w1pk, xz, M1, 384, DM);
  }
  // 2) depthwise conv + SiLU
  n = B_ * L_ * DIN;
  k_conv_silu<<<(n + 255) / 256, 256, 0, stream>>>(xz, conv_w, conv_b, xconv, xsp);
  // 3) per-direction dt/B/C projections: 4x (18432x48)=(18432x192)*(192x48) [WMMA, NT=3]
  for (int k = 0; k < KK; ++k) {
    int waves = (M1 / 16) * ((CPAD / 16) / 3);
    int thr = waves * 32;
    gemm_bf16_wmma<3><<<(thr + 255) / 256, 256, 0, stream>>>(
        xsp, wxpk + (size_t)k * CPAD * DIN, xdbl + (size_t)k * M1 * CPAD, M1, CPAD, DIN);
  }
  // 4) selective scan: 32 blocks (b,k) x 192 threads (d), serial over L=2304
  k_scan<<<B_ * KK, DIN, 0, stream>>>(xconv, xdbl, dt_w, dt_b, A_logs, Ds, ys);
  // 5) merge 4 directions + LayerNorm + SiLU gate
  k_merge_ln_gate<<<M1, DIN, 0, stream>>>(ys, xz, ln_w, ln_b, yg);
  // 6) out_proj: out(18432x96) = yg(18432x192) * W^T(192x96)  [WMMA, NT=3]
  {
    int waves = (M1 / 16) * ((DM / 16) / 3);
    int thr = waves * 32;
    gemm_bf16_wmma<3><<<(thr + 255) / 256, 256, 0, stream>>>(yg, wopk, out, M1, DM, DIN);
  }
}
